// Mamba_Layer_60292750901834
// MI455X (gfx1250) — compile-verified, matrix operand order checked by reading, not run on hardware
//
#include <hip/hip_runtime.h>

// ---------------- Mamba layer for gfx1250 (wave32, WMMA bf16, async-LDS) -----
#define B_SZ   4
#define L_SEQ  2048
#define DM     256
#define DI     512
#define DST    16
#define RK     16
#define MTOT   (B_SZ * L_SEQ)          // 8192 rows
#define LN_EPS 1e-5f

typedef __attribute__((ext_vector_type(16))) __bf16 v16bf;
typedef __attribute__((ext_vector_type(8)))  float  v8f;
typedef int v4i __attribute__((vector_size(16)));    // matches builtin pointee

#define AS_GLOBAL __attribute__((address_space(1)))
#define AS_SHARED __attribute__((address_space(3)))

// ---- f32 -> bf16 (round to nearest even) ----
__device__ __forceinline__ unsigned short f2bf(float f) {
    unsigned int u = __float_as_uint(f);
    u += 0x7FFFu + ((u >> 16) & 1u);
    return (unsigned short)(u >> 16);
}

// Load one 16-bf16-per-lane WMMA fragment: two 16B chunks at p and p+16 elems.
__device__ __forceinline__ v16bf load_frag(const unsigned short* p) {
    union { v16bf v; uint4 q[2]; } f;
    f.q[0] = *(const uint4*)(p);
    f.q[1] = *(const uint4*)(p + 16);
    return f.v;
}

// ---------------- elementwise converters ----------------
__global__ void k_f32_to_bf16(const float* __restrict__ src,
                              unsigned short* __restrict__ dst, int n) {
    int i = blockIdx.x * blockDim.x + threadIdx.x;
    if (i < n) dst[i] = f2bf(src[i]);
}

// dt_proj_w [512,16] -> bf16 [512,32] zero padded in K
__global__ void k_pad_dtw(const float* __restrict__ w,
                          unsigned short* __restrict__ dst) {
    int i = blockIdx.x * blockDim.x + threadIdx.x;       // 512*32
    if (i >= DI * 32) return;
    int row = i >> 5, c = i & 31;
    float v = (c < RK) ? w[row * RK + c] : 0.f;
    dst[i] = f2bf(v);
}

// x_proj_w [48,512] -> bf16 [64,512] zero padded in N (rows)
__global__ void k_pad_xpw(const float* __restrict__ w,
                          unsigned short* __restrict__ dst) {
    int i = blockIdx.x * blockDim.x + threadIdx.x;       // 64*512
    if (i >= 64 * DI) return;
    int row = i >> 9, c = i & (DI - 1);
    float v = (row < 48) ? w[row * DI + c] : 0.f;
    dst[i] = f2bf(v);
}

// ---------------- WMMA GEMM: C[M,N] = A[M,K] * W[N,K]^T ----------------
// A, W bf16 row-major (ld = K), C f32 row-major (ld = N). N, K multiples of 64/32.
// block = (32,4): 4 waves, 128(M) x 64(N) per block, 32x64 per wave.
// Weight strip chunks (64 rows x 32 k = 4 KB) staged to LDS with async
// loads (ASYNCcnt), double-buffered so the fill overlaps the WMMAs.
__global__ __launch_bounds__(128)
void k_gemm_bf16(const unsigned short* __restrict__ A,
                 const unsigned short* __restrict__ W,
                 float* __restrict__ C, int M, int N, int K) {
    __shared__ unsigned short wch[2][64 * 32];           // 2 x 4 KB
    const int lane = threadIdx.x;
    const int wv   = threadIdx.y;
    const int tid  = wv * 32 + lane;
    const int nb   = blockIdx.x * 64;
    const int mw   = blockIdx.y * 128 + wv * 32;
    const int r    = lane & 15;
    const int koff = (lane >> 4) << 3;                   // ISA 16-bit A/B layout

    v8f acc[2][4];
#pragma unroll
    for (int i = 0; i < 2; ++i)
#pragma unroll
        for (int t = 0; t < 4; ++t)
            acc[i][t] = (v8f){0.f,0.f,0.f,0.f,0.f,0.f,0.f,0.f};

    const unsigned short* arow0 = A + (size_t)(mw + r) * K;
    const unsigned short* arow1 = A + (size_t)(mw + 16 + r) * K;

    // stage one 64x32 weight chunk (4096 B) into LDS buffer `bufsel`
    auto stage = [&](int k0, int bufsel) {
        int f = tid * 16;                                // this thread's bytes
#pragma unroll
        for (int h = 0; h < 2; ++h, f += 2048) {
            const int row    = f >> 6;                   // 64 B per row
            const int within = f & 63;
            const char* g = (const char*)(W + (size_t)(nb + row) * K + k0) + within;
            char*       l = (char*)(&wch[bufsel][0]) + f;
            __builtin_amdgcn_global_load_async_to_lds_b128(
                (AS_GLOBAL v4i*)g, (AS_SHARED v4i*)l, 0, 0);
        }
    };

    const int nchunks = K >> 5;
    stage(0, 0);
    for (int kc = 0; kc < nchunks; ++kc) {
        const int k0 = kc << 5;
        __builtin_amdgcn_s_wait_asynccnt(0);             // my chunk landed
        __syncthreads();                                 // everyone's landed
        if (kc + 1 < nchunks) stage(k0 + 32, (kc + 1) & 1);

        v16bf a0 = load_frag(arow0 + k0 + koff);
        v16bf a1 = load_frag(arow1 + k0 + koff);
        const unsigned short* wbase = &wch[kc & 1][0];
#pragma unroll
        for (int t = 0; t < 4; ++t) {
            v16bf bfr = load_frag(wbase + (t * 16 + r) * 32 + koff);
            acc[0][t] = __builtin_amdgcn_wmma_f32_16x16x32_bf16(
                            false, a0, false, bfr, (short)0, acc[0][t], false, false);
            acc[1][t] = __builtin_amdgcn_wmma_f32_16x16x32_bf16(
                            false, a1, false, bfr, (short)0, acc[1][t], false, false);
        }
        __syncthreads();                                 // done reading this buf
    }

    // D layout: VGPR rr -> row mtile + rr + 8*(lane>=16), col = n0 + (lane&15)
#pragma unroll
    for (int i = 0; i < 2; ++i) {
        const int mb = mw + i * 16 + ((lane >> 4) << 3);
#pragma unroll
        for (int t = 0; t < 4; ++t) {
            const int col = nb + t * 16 + r;
#pragma unroll
            for (int rr = 0; rr < 8; ++rr)
                C[(size_t)(mb + rr) * N + col] = acc[i][t][rr];
        }
    }
}

// ---------------- causal depthwise conv (4 taps) + SiLU ----------------
__global__ void k_conv_silu(const float* __restrict__ xz,
                            const float* __restrict__ cw,
                            const float* __restrict__ cb,
                            float* __restrict__ xt,
                            unsigned short* __restrict__ xtb) {
    int idx = blockIdx.x * blockDim.x + threadIdx.x;   // MTOT*DI
    if (idx >= MTOT * DI) return;
    int e = idx & (DI - 1);
    int l = (idx >> 9) & (L_SEQ - 1);
    int b = idx >> 20;
    float s = cb[e];
#pragma unroll
    for (int j = 0; j < 4; ++j) {
        int ls = l - 3 + j;
        if (ls >= 0)
            s += xz[(((size_t)(b << 11) + ls) << 10) + e] * cw[e * 4 + j];
    }
    float v = s / (1.f + __expf(-s));
    xt[idx]  = v;
    xtb[idx] = f2bf(v);
}

// dbc [8192,64] -> dt bf16 [8192,32] (zero-padded)
__global__ void k_make_dtb(const float* __restrict__ dbc,
                           unsigned short* __restrict__ dtb) {
    int i = blockIdx.x * blockDim.x + threadIdx.x;     // MTOT*32
    if (i >= MTOT * 32) return;
    int m = i >> 5, c = i & 31;
    float v = (c < RK) ? dbc[m * 64 + c] : 0.f;
    dtb[i] = f2bf(v);
}

// delta = softplus(delta_pre + dt_proj_b), in place
__global__ void k_softplus(float* __restrict__ d, const float* __restrict__ bias) {
    int i = blockIdx.x * blockDim.x + threadIdx.x;     // MTOT*DI
    if (i >= MTOT * DI) return;
    float v = d[i] + bias[i & (DI - 1)];
    d[i] = (v > 20.f) ? v : log1pf(__expf(v));
}

// ---------------- selective scan + skip + gate ----------------
// One wave32 handles 2 channels (e0, e0+1); lane&15 = state index n.
__global__ void k_scan(const float* __restrict__ delta,
                       const float* __restrict__ xt,
                       const float* __restrict__ dbc,
                       const float* __restrict__ xz,
                       const float* __restrict__ A_log,
                       const float* __restrict__ Dp,
                       unsigned short* __restrict__ yb) {
    const int lane = threadIdx.x;
    const int w    = blockIdx.x * blockDim.y + threadIdx.y;   // 0..1023
    const int c0   = w * 2;
    const int b    = c0 >> 9;
    const int e    = (c0 & (DI - 1)) + (lane >> 4);
    const int n    = lane & 15;

    const float a     = -__expf(A_log[e * DST + n]);          // loop invariant
    const float dskip = Dp[e];
    float h = 0.f;

    for (int l = 0; l < L_SEQ; ++l) {
        const int bl = (b << 11) + l;
        if ((l & 63) == 0 && l + 64 < L_SEQ) {                // global_prefetch_b8
            __builtin_prefetch(&delta[((bl + 64) << 9) + e], 0, 1);
            __builtin_prefetch(&dbc[((bl + 64) << 6) + 16 + n], 0, 1);
        }
        const float dlt = delta[(bl << 9) + e];
        const float u   = xt[(bl << 9) + e];
        const float Bn  = dbc[(bl << 6) + 16 + n];
        const float Cn  = dbc[(bl << 6) + 32 + n];
        const float dA  = __expf(dlt * a);
        h = fmaf(dA, h, dlt * u * Bn);
        float t = h * Cn;
        t += __shfl_xor(t, 1);
        t += __shfl_xor(t, 2);
        t += __shfl_xor(t, 4);
        t += __shfl_xor(t, 8);                                 // 16-lane sum
        if (n == 0) {
            const float z = xz[(bl << 10) + DI + e];
            float y = t + u * dskip;
            y *= z / (1.f + __expf(-z));                       // * silu(z)
            yb[(bl << 9) + e] = f2bf(y);
        }
    }
}

// ---------------- residual + LayerNorm + transpose to [B, DM, L] ----------------
__global__ void k_ln_transpose(const float* __restrict__ x,
                               const float* __restrict__ mo,
                               const float* __restrict__ lnw,
                               const float* __restrict__ lnb,
                               float* __restrict__ out) {
    __shared__ float tile[DM * 33];
    const int blk  = blockIdx.x;
    const int b    = blk >> 6;                // 64 blocks per batch
    const int l0   = (blk & 63) << 5;
    const int lane = threadIdx.x;
    const int wv   = threadIdx.y;

    for (int rr = 0; rr < 4; ++rr) {
        const int ll = wv * 4 + rr;
        const size_t rowoff = ((size_t)((b << 11) + l0 + ll)) << 8;   // *256
        const float4* xr = (const float4*)(x  + rowoff);
        const float4* mr = (const float4*)(mo + rowoff);
        float4 p0 = xr[lane * 2], p1 = xr[lane * 2 + 1];
        float4 q0 = mr[lane * 2], q1 = mr[lane * 2 + 1];
        float v[8] = { p0.x + q0.x, p0.y + q0.y, p0.z + q0.z, p0.w + q0.w,
                       p1.x + q1.x, p1.y + q1.y, p1.z + q1.z, p1.w + q1.w };
        float s = 0.f, sq = 0.f;
#pragma unroll
        for (int j = 0; j < 8; ++j) { s += v[j]; sq += v[j] * v[j]; }
#pragma unroll
        for (int off = 1; off < 32; off <<= 1) {
            s  += __shfl_xor(s,  off);
            sq += __shfl_xor(sq, off);
        }
        const float mu  = s * (1.f / DM);
        const float var = sq * (1.f / DM) - mu * mu;
        const float inv = rsqrtf(var + LN_EPS);
#pragma unroll
        for (int j = 0; j < 8; ++j) {
            const int d = lane * 8 + j;
            tile[d * 33 + ll] = (v[j] - mu) * inv * lnw[d] + lnb[d];
        }
    }
    __syncthreads();
    const int tid = wv * 32 + lane;
    for (int idx = tid; idx < DM * 32; idx += 256) {
        const int d = idx >> 5, ll = idx & 31;
        out[(((size_t)(b << 8) + d) << 11) + l0 + ll] = tile[d * 33 + ll];
    }
}

// ---------------- host launch ----------------
extern "C" void kernel_launch(void* const* d_in, const int* in_sizes, int n_in,
                              void* d_out, int out_size, void* d_ws, size_t ws_size,
                              hipStream_t stream) {
    const float* x         = (const float*)d_in[0];
    const float* in_proj_w = (const float*)d_in[1];
    const float* conv_w    = (const float*)d_in[2];
    const float* conv_b    = (const float*)d_in[3];
    const float* x_proj_w  = (const float*)d_in[4];
    const float* dt_proj_w = (const float*)d_in[5];
    const float* dt_proj_b = (const float*)d_in[6];
    const float* A_log     = (const float*)d_in[7];
    const float* D_param   = (const float*)d_in[8];
    const float* out_proj_w= (const float*)d_in[9];
    const float* ln_w      = (const float*)d_in[10];
    const float* ln_b      = (const float*)d_in[11];
    float* out = (float*)d_out;

    char* ws = (char*)d_ws;
    size_t o = 0;
    auto take = [&](size_t bytes) -> char* {
        char* r = ws + o; o += (bytes + 255) & ~(size_t)255; return r;
    };
    float*          xz     = (float*)take((size_t)MTOT * 1024 * 4);   // 32 MB
    float*          xt     = (float*)take((size_t)MTOT * DI * 4);     // 16 MB
    float*          delta  = (float*)take((size_t)MTOT * DI * 4);     // 16 MB
    float*          dbc    = (float*)take((size_t)MTOT * 64 * 4);     // padded N=64
    float*          mout   = (float*)take((size_t)MTOT * DM * 4);
    unsigned short* xb     = (unsigned short*)take((size_t)MTOT * DM * 2);
    unsigned short* xtb    = (unsigned short*)take((size_t)MTOT * DI * 2);
    unsigned short* dtb    = (unsigned short*)take((size_t)MTOT * 32 * 2);
    unsigned short* ybuf   = (unsigned short*)take((size_t)MTOT * DI * 2);
    unsigned short* w_inb  = (unsigned short*)take((size_t)1024 * DM * 2);
    unsigned short* w_xpb  = (unsigned short*)take((size_t)64 * DI * 2);  // padded
    unsigned short* w_dtb  = (unsigned short*)take((size_t)DI * 32 * 2);  // padded
    unsigned short* w_outb = (unsigned short*)take((size_t)DM * DI * 2);

    auto cvt = [&](const float* s, unsigned short* d, int n) {
        k_f32_to_bf16<<<(n + 255) / 256, 256, 0, stream>>>(s, d, n);
    };
    cvt(x, xb, MTOT * DM);
    cvt(in_proj_w, w_inb, 1024 * DM);
    cvt(out_proj_w, w_outb, DM * DI);
    k_pad_xpw<<<(64 * DI + 255) / 256, 256, 0, stream>>>(x_proj_w, w_xpb);
    k_pad_dtw<<<(DI * 32 + 255) / 256, 256, 0, stream>>>(dt_proj_w, w_dtb);

    dim3 gblk(32, 4);
    // in_proj: [8192,1024] = xb[8192,256] * in_proj_w[1024,256]^T
    k_gemm_bf16<<<dim3(1024 / 64, MTOT / 128), gblk, 0, stream>>>(xb, w_inb, xz, MTOT, 1024, DM);
    // conv + SiLU
    k_conv_silu<<<(MTOT * DI + 255) / 256, 256, 0, stream>>>(xz, conv_w, conv_b, xt, xtb);
    // x_proj (N padded to 64): dbc[8192,64] = xtb * w_xpb^T
    k_gemm_bf16<<<dim3(1, MTOT / 128), gblk, 0, stream>>>(xtb, w_xpb, dbc, MTOT, 64, DI);
    k_make_dtb<<<(MTOT * 32 + 255) / 256, 256, 0, stream>>>(dbc, dtb);
    // dt_proj: [8192,512] = dtb[8192,32] * dt_proj_w[512,32]^T
    k_gemm_bf16<<<dim3(DI / 64, MTOT / 128), gblk, 0, stream>>>(dtb, w_dtb, delta, MTOT, DI, 32);
    k_softplus<<<(MTOT * DI + 255) / 256, 256, 0, stream>>>(delta, dt_proj_b);
    // selective scan fused with skip + gate -> bf16
    k_scan<<<256, dim3(32, 4), 0, stream>>>(delta, xt, dbc, xz, A_log, D_param, ybuf);
    // out_proj: [8192,256] = ybuf * out_proj_w^T
    k_gemm_bf16<<<dim3(DM / 64, MTOT / 128), gblk, 0, stream>>>(ybuf, w_outb, mout, MTOT, DM, DI);
    // residual + LN + transpose
    k_ln_transpose<<<256, dim3(32, 8), 0, stream>>>(x, mout, ln_w, ln_b, out);
}
